// RBF_5549097746661
// MI455X (gfx1250) — matrix operand order repaired and checked by reading, not compiled, not run on hardware
//
#include <hip/hip_runtime.h>

// ---------------------------------------------------------------------------
// RBF kernel + bandwidth-median + gradient for MI455X (gfx1250, wave32, WMMA)
//   N = 8192 rows, D = 256 features
//   out = [ K (8192x8192 f32) | grad_K (8192x256 f32) ]
// Strategy: pre-split X/Y into f16 (hi,lo) pairs once (L2-resident), so both
// GEMMs run as pure load_b128 + v_wmma_f32_16x16x32_f16 streams (3 WMMAs per
// tile give ~fp32 accuracy).  Exact median via 2-pass 16-bit radix select on
// float bit patterns.  exp+column-sum fused in one in-place pass over K.
// ---------------------------------------------------------------------------

typedef __attribute__((ext_vector_type(16))) _Float16 v16h;
typedef __attribute__((ext_vector_type(8)))  _Float16 v8h;
typedef __attribute__((ext_vector_type(8)))  float    v8f;
typedef __attribute__((ext_vector_type(4)))  float    v4f;

#define NROW  8192
#define DDIM  256
#define KTOT  67108864ull            // 8192*8192
// off-diagonal count = 8192*8191 = 67,100,672 (even) -> median = avg of ranks:
#define RANK0 33550335u
#define RANK1 33550336u

// ---------------- helpers ----------------------------------------------------
__device__ __forceinline__ void cvt_split16(const float* f, v16h& hi, v16h& lo) {
#pragma unroll
  for (int t = 0; t < 16; ++t) {
    _Float16 h = (_Float16)f[t];
    hi[t] = h;
    lo[t] = (_Float16)(f[t] - (float)h);
  }
}

// A-tile fragment: 8 halves at p, 8 halves at p+16 (k-chunks kb, kb+16)
__device__ __forceinline__ v16h load_h16_a(const _Float16* p) {
  union { v16h v; v8h h[2]; } u;
  u.h[0] = *(const v8h*)(p);
  u.h[1] = *(const v8h*)(p + 16);
  return u.v;
}
// B-tile fragment: 16 contiguous halves at p
__device__ __forceinline__ v16h load_h16_b(const _Float16* p) {
  union { v16h v; v8h h[2]; } u;
  u.h[0] = *(const v8h*)(p);
  u.h[1] = *(const v8h*)(p + 8);
  return u.v;
}

// split-precision product: C += Ah*Bh + Ah*Bl + Al*Bh  (~fp32-accurate)
__device__ __forceinline__ v8f wmma3(v16h ah, v16h al, v16h bh, v16h bl, v8f c) {
  c = __builtin_amdgcn_wmma_f32_16x16x32_f16(false, ah, false, bh, (short)0, c, false, false);
  c = __builtin_amdgcn_wmma_f32_16x16x32_f16(false, ah, false, bl, (short)0, c, false, false);
  c = __builtin_amdgcn_wmma_f32_16x16x32_f16(false, al, false, bh, (short)0, c, false, false);
  return c;
}

// ---------------- 0) zero accumulator region of ws ---------------------------
__global__ void zero_kernel(float* p, int n) {
  int i = blockIdx.x * blockDim.x + threadIdx.x;
  if (i < n) p[i] = 0.0f;
}

// ---------------- 1) row norms XX, YY ---------------------------------------
__global__ __launch_bounds__(256) void rownorm_kernel(const float* __restrict__ X,
                                                      const float* __restrict__ Y,
                                                      float* __restrict__ XX,
                                                      float* __restrict__ YY) {
  int gw = (blockIdx.x * 256 + threadIdx.x) >> 5;  // one wave32 per row
  int lane = threadIdx.x & 31;
  const float* src;
  float* dst;
  int row;
  if (gw < NROW) { src = X; dst = XX; row = gw; }
  else           { src = Y; dst = YY; row = gw - NROW; }
  const v4f* p = (const v4f*)(src + (size_t)row * DDIM);
  v4f a = p[lane], b = p[lane + 32];
  float s = a.x*a.x + a.y*a.y + a.z*a.z + a.w*a.w
          + b.x*b.x + b.y*b.y + b.z*b.z + b.w*b.w;
#pragma unroll
  for (int off = 16; off > 0; off >>= 1) s += __shfl_xor(s, off, 32);
  if (lane == 0) dst[row] = s;
}

// ---------------- 2a) pre-split X,Y -> f16 hi/lo arrays ----------------------
__global__ __launch_bounds__(256) void split_xy_kernel(const float* __restrict__ X,
                                                       const float* __restrict__ Y,
                                                       _Float16* __restrict__ Xh,
                                                       _Float16* __restrict__ Xl,
                                                       _Float16* __restrict__ Yh,
                                                       _Float16* __restrict__ Yl) {
  size_t idx = (size_t)blockIdx.x * 256 + threadIdx.x;   // 2,097,152 elements
  float x = X[idx];
  _Float16 xh = (_Float16)x;
  Xh[idx] = xh;
  Xl[idx] = (_Float16)(x - (float)xh);
  float y = Y[idx];
  _Float16 yh = (_Float16)y;
  Yh[idx] = yh;
  Yl[idx] = (_Float16)(y - (float)yh);
}

// ---------------- 2b) XT hi/lo = split(X^T)  (grad-GEMM B operand) -----------
__global__ __launch_bounds__(256) void transpose_split_kernel(const float* __restrict__ X,
                                                              _Float16* __restrict__ XTh,
                                                              _Float16* __restrict__ XTl) {
  __shared__ float tile[64][65];
  int bx = blockIdx.x & 3;      // d-block (256/64)
  int by = blockIdx.x >> 2;     // i-block (8192/64)
  int i0 = by * 64, d0 = bx * 64;
#pragma unroll
  for (int it = 0; it < 16; ++it) {
    int idx = it * 256 + threadIdx.x;
    int r = idx >> 6, c = idx & 63;                   // r = i-local, c = d-local
    tile[c][r] = X[(size_t)(i0 + r) * DDIM + d0 + c]; // coalesced read
  }
  __syncthreads();
#pragma unroll
  for (int it = 0; it < 16; ++it) {
    int idx = it * 256 + threadIdx.x;
    int r = idx >> 6, c = idx & 63;                   // r = d-local, c = i-local
    float v = tile[r][c];
    _Float16 h = (_Float16)v;
    size_t o = (size_t)(d0 + r) * NROW + i0 + c;      // coalesced write
    XTh[o] = h;
    XTl[o] = (_Float16)(v - (float)h);
  }
}

// ---------------- 3) pd = XX + YY^T - 2 X Y^T  (split-f16 WMMA) --------------
// One wave computes a 64x64 block: 4x4 tiles of 16x16, 3 WMMAs per tile per
// k-step.  B tiles (4 pairs) stay live; A tiles streamed per mt to keep the
// wave under 256 VGPRs (no s_set_vgpr_msb churn).
__global__ __launch_bounds__(256) void gemm_pd_kernel(const _Float16* __restrict__ Xh,
                                                      const _Float16* __restrict__ Xl,
                                                      const _Float16* __restrict__ Yh,
                                                      const _Float16* __restrict__ Yl,
                                                      const float* __restrict__ XX,
                                                      const float* __restrict__ YY,
                                                      float* __restrict__ PD) {
  int gw   = (blockIdx.x * 256 + threadIdx.x) >> 5;  // 16384 wave jobs
  int lane = threadIdx.x & 31;
  int m0 = (gw >> 7) * 64;
  int j0 = (gw & 127) * 64;

  v8f zero = {0.f, 0.f, 0.f, 0.f, 0.f, 0.f, 0.f, 0.f};
  v8f c[4][4];
#pragma unroll
  for (int mt = 0; mt < 4; ++mt)
#pragma unroll
    for (int nt = 0; nt < 4; ++nt) c[mt][nt] = zero;

  int kbA = (lane >> 4) * 16;   // A k-chunk base
  int kbB = (lane >> 4) * 16;   // B k-chunk base
  int lr  = lane & 15;

  for (int k0 = 0; k0 < DDIM; k0 += 32) {
    v16h bh[4], bl[4];
#pragma unroll
    for (int nt = 0; nt < 4; ++nt) {
      size_t off = (size_t)(j0 + 16 * nt + lr) * DDIM + k0 + kbB;
      bh[nt] = load_h16_b(Yh + off);
      bl[nt] = load_h16_b(Yl + off);
    }
#pragma unroll
    for (int mt = 0; mt < 4; ++mt) {
      size_t off = (size_t)(m0 + 16 * mt + lr) * DDIM + k0 + kbA;
      v16h ah = load_h16_b(Xh + off);
      v16h al = load_h16_b(Xl + off);
#pragma unroll
      for (int nt = 0; nt < 4; ++nt)
        c[mt][nt] = wmma3(ah, al, bh[nt], bl[nt], c[mt][nt]);
    }
  }

  // epilogue: pd = XX[i] + YY[j] - 2*dot
  int rhi = 8 * (lane >> 4);
#pragma unroll
  for (int nt = 0; nt < 4; ++nt) {
    int j = j0 + 16 * nt + lr;
    float yy = YY[j];
#pragma unroll
    for (int mt = 0; mt < 4; ++mt) {
#pragma unroll
      for (int r = 0; r < 8; ++r) {
        int i = m0 + 16 * mt + r + rhi;
        PD[(size_t)i * NROW + j] = XX[i] + yy - 2.0f * c[mt][nt][r];
      }
    }
  }
}

// ---------------- 4) radix-select pass 1: top-16-bit histogram ---------------
__global__ __launch_bounds__(1024) void hist1_kernel(const float* __restrict__ PD,
                                                     unsigned* __restrict__ hist) {
  extern __shared__ unsigned sh[];                 // 65536 bins (256KB LDS)
  for (int i = threadIdx.x; i < 65536; i += 1024) sh[i] = 0u;
  __syncthreads();
  size_t stride = (size_t)gridDim.x * 1024;
  for (size_t idx = (size_t)blockIdx.x * 1024 + threadIdx.x; idx < KTOT; idx += stride) {
    unsigned u = __float_as_uint(fmaxf(PD[idx], 0.0f));
    atomicAdd(&sh[u >> 16], 1u);
  }
  __syncthreads();
  for (int i = threadIdx.x; i < 65536; i += 1024)
    if (sh[i]) atomicAdd(&hist[i], sh[i]);
}

// remove diagonal entries (median is over off-diagonal only)
__global__ void diag1_kernel(const float* __restrict__ PD, unsigned* __restrict__ hist) {
  int i = blockIdx.x * blockDim.x + threadIdx.x;
  if (i < NROW) {
    unsigned u = __float_as_uint(fmaxf(PD[(size_t)i * (NROW + 1)], 0.0f));
    atomicSub(&hist[u >> 16], 1u);
  }
}

// locate top-16-bit bins holding ranks RANK0/RANK1
__global__ __launch_bounds__(1024) void select1_kernel(const unsigned* __restrict__ hist,
                                                       unsigned* __restrict__ selu) {
  __shared__ unsigned ps[1024];
  __shared__ unsigned fbin[2], frank[2];
  int tid = threadIdx.x, base = tid * 64;
  unsigned s = 0;
  for (int b = 0; b < 64; ++b) s += hist[base + b];
  ps[tid] = s;
  __syncthreads();
  if (tid == 0) {
    unsigned run = 0;
    for (int k = 0; k < 1024; ++k) { unsigned t = ps[k]; ps[k] = run; run += t; }
  }
  __syncthreads();
  unsigned cum = ps[tid];
  for (int b = 0; b < 64; ++b) {
    unsigned cnt = hist[base + b];
    if (RANK0 >= cum && RANK0 < cum + cnt) { fbin[0] = base + b; frank[0] = RANK0 - cum; }
    if (RANK1 >= cum && RANK1 < cum + cnt) { fbin[1] = base + b; frank[1] = RANK1 - cum; }
    cum += cnt;
  }
  __syncthreads();
  if (tid == 0) { selu[0] = fbin[0]; selu[1] = fbin[1]; selu[2] = frank[0]; selu[3] = frank[1]; }
}

// ---------------- 5) radix-select pass 2: low-16-bit histograms --------------
__global__ __launch_bounds__(256) void hist2_kernel(const float* __restrict__ PD,
                                                    const unsigned* __restrict__ selu,
                                                    unsigned* __restrict__ hist2) {
  unsigned binA = selu[0], binB = selu[1];
  size_t stride = (size_t)gridDim.x * blockDim.x;
  for (size_t idx = (size_t)blockIdx.x * blockDim.x + threadIdx.x; idx < KTOT; idx += stride) {
    unsigned u = __float_as_uint(fmaxf(PD[idx], 0.0f));
    unsigned t = u >> 16;
    if (t == binA) atomicAdd(&hist2[u & 0xFFFFu], 1u);
    if (t == binB) atomicAdd(&hist2[65536 + (u & 0xFFFFu)], 1u);
  }
}

__global__ void diag2_kernel(const float* __restrict__ PD,
                             const unsigned* __restrict__ selu,
                             unsigned* __restrict__ hist2) {
  int i = blockIdx.x * blockDim.x + threadIdx.x;
  if (i < NROW) {
    unsigned u = __float_as_uint(fmaxf(PD[(size_t)i * (NROW + 1)], 0.0f));
    unsigned t = u >> 16;
    if (t == selu[0]) atomicSub(&hist2[u & 0xFFFFu], 1u);
    if (t == selu[1]) atomicSub(&hist2[65536 + (u & 0xFFFFu)], 1u);
  }
}

// finish selection: exact median -> sigma -> gamma
__global__ __launch_bounds__(1024) void select2_kernel(const unsigned* __restrict__ hist2,
                                                       const unsigned* __restrict__ selu_in,
                                                       float* __restrict__ self) {
  __shared__ unsigned ps[1024];
  __shared__ unsigned flo[2];
  int tid = threadIdx.x, base = tid * 64;
  for (int hsel = 0; hsel < 2; ++hsel) {
    const unsigned* hh = hist2 + hsel * 65536;
    unsigned rank = selu_in[2 + hsel];
    unsigned s = 0;
    for (int b = 0; b < 64; ++b) s += hh[base + b];
    ps[tid] = s;
    __syncthreads();
    if (tid == 0) {
      unsigned run = 0;
      for (int k = 0; k < 1024; ++k) { unsigned t = ps[k]; ps[k] = run; run += t; }
    }
    __syncthreads();
    unsigned cum = ps[tid];
    for (int b = 0; b < 64; ++b) {
      unsigned cnt = hh[base + b];
      if (rank >= cum && rank < cum + cnt) flo[hsel] = (unsigned)(base + b);
      cum += cnt;
    }
    __syncthreads();
  }
  if (tid == 0) {
    float medA = __uint_as_float((selu_in[0] << 16) | flo[0]);
    float medB = __uint_as_float((selu_in[1] << 16) | flo[1]);
    float h = 0.5f * (medA + medB);
    float sigma = sqrtf(h / logf((float)(NROW + 1)));
    sigma = fmaxf(sigma, 0.01f);
    float gamma = 1.0f / (2.0f * sigma * sigma);
    self[4] = gamma;
    self[5] = -2.0f * gamma;
  }
}

// ---------------- 6) K = exp(-gamma*pd) in place + fused column sums ---------
__global__ __launch_bounds__(256) void exp_colsum_kernel(float* __restrict__ Km,
                                                         const float* __restrict__ self,
                                                         float* __restrict__ colsum) {
  float gamma = self[4];
  int t = threadIdx.x;
  int rb = blockIdx.x * 64;
  float acc[32];
#pragma unroll
  for (int q = 0; q < 32; ++q) acc[q] = 0.0f;
  for (int r = 0; r < 64; ++r) {
    size_t base = (size_t)(rb + r) * NROW;
#pragma unroll
    for (int q = 0; q < 32; ++q) {
      int j = t + 256 * q;
      float k = __expf(-gamma * Km[base + j]);
      Km[base + j] = k;
      acc[q] += k;
    }
  }
#pragma unroll
  for (int q = 0; q < 32; ++q) atomicAdd(&colsum[t + 256 * q], acc[q]);
}

// ---------------- 7) grad = -2g*(K^T X - Y*colsum)  (split-f16 WMMA) ---------
// WG = 4 waves; block computes 64 rows(j) x 256 cols(d).  A = K^T via padded-LDS
// transpose of 32x64 K blocks (split in-register, 1 tile/wave/k-step);
// B = pre-split rows of X^T (contiguous k, pure load_b128).
__global__ __launch_bounds__(128) void grad_gemm_kernel(const float* __restrict__ Km,
                                                        const _Float16* __restrict__ XTh,
                                                        const _Float16* __restrict__ XTl,
                                                        const float* __restrict__ Y,
                                                        const float* __restrict__ colsum,
                                                        const float* __restrict__ self,
                                                        float* __restrict__ G) {
  __shared__ float kt[64 * 36];   // kt[j_local*36 + i_local], padded, 16B-aligned rows
  int lane = threadIdx.x & 31;
  int wv   = threadIdx.x >> 5;
  int j0   = blockIdx.x * 64;
  int lr   = lane & 15;
  int kbA  = (lane >> 4) * 16;
  int kbB  = (lane >> 4) * 16;

  v8f zero = {0.f, 0.f, 0.f, 0.f, 0.f, 0.f, 0.f, 0.f};
  v8f acc[16];
#pragma unroll
  for (int nt = 0; nt < 16; ++nt) acc[nt] = zero;

  for (int i0 = 0; i0 < NROW; i0 += 32) {
    __syncthreads();
    // cooperative load K[i0..i0+31][j0..j0+63] -> transposed LDS
#pragma unroll
    for (int it = 0; it < 16; ++it) {
      int idx = it * 128 + threadIdx.x;
      int r = idx >> 6, col = idx & 63;                      // r = i-local, col = j-local
      kt[col * 36 + r] = Km[(size_t)(i0 + r) * NROW + j0 + col];
    }
    __syncthreads();

    // A tile for this wave: rows j_local in [16*wv, 16*wv+16)
    int jl = 16 * wv + lr;
    float fa[16];
#pragma unroll
    for (int tt = 0; tt < 8; ++tt) {
      fa[tt]     = kt[jl * 36 + kbA + tt];
      fa[8 + tt] = kt[jl * 36 + kbA + 8 + tt];
    }
    v16h ah, al;
    cvt_split16(fa, ah, al);

#pragma unroll
    for (int nt = 0; nt < 16; ++nt) {
      size_t off = (size_t)(16 * nt + lr) * NROW + i0 + kbB;
      v16h bh = load_h16_b(XTh + off);
      v16h bl = load_h16_b(XTl + off);
      acc[nt] = wmma3(ah, al, bh, bl, acc[nt]);
    }
  }

  // epilogue: grad = m2g * (G - Y[j][d] * colsum[j])
  float m2g = self[5];
  int rhi = 8 * (lane >> 4);
#pragma unroll
  for (int nt = 0; nt < 16; ++nt) {
    int d = 16 * nt + lr;
#pragma unroll
    for (int r = 0; r < 8; ++r) {
      int j = j0 + 16 * wv + r + rhi;
      float v = m2g * (acc[nt][r] - Y[(size_t)j * DDIM + d] * colsum[j]);
      G[(size_t)j * DDIM + d] = v;
    }
  }
}

// ---------------------------------------------------------------------------
extern "C" void kernel_launch(void* const* d_in, const int* in_sizes, int n_in,
                              void* d_out, int out_size, void* d_ws, size_t ws_size,
                              hipStream_t stream) {
  const float* X = (const float*)d_in[0];
  const float* Y = (const float*)d_in[1];
  float* out = (float*)d_out;
  float* PD  = out;                 // 67,108,864 floats: pd, then K (in place)
  float* G   = out + KTOT;          // 2,097,152 floats: grad_K

  // workspace layout (floats):
  float*    ws      = (float*)d_ws;
  float*    XX      = ws;                       // 8192
  float*    YY      = ws + 8192;                // 8192
  float*    colsum  = ws + 16384;               // 8192        (zeroed)
  unsigned* hist1   = (unsigned*)(ws + 24576);  // 65536       (zeroed)
  unsigned* hist2   = hist1 + 65536;            // 131072      (zeroed)
  unsigned* selu    = hist2 + 131072;           // 64          (zeroed)
  float*    self    = (float*)selu;
  // half-precision split arrays at ws + 1MB (each 2,097,152 halves = 4MB)
  _Float16* hb  = (_Float16*)(ws + 262144);
  _Float16* Xh  = hb;
  _Float16* Xl  = hb + 1 * 2097152;
  _Float16* Yh  = hb + 2 * 2097152;
  _Float16* Yl  = hb + 3 * 2097152;
  _Float16* XTh = hb + 4 * 2097152;
  _Float16* XTl = hb + 5 * 2097152;

  // 0) zero accumulators: colsum + hist1 + hist2 + sel = 204,864 floats
  zero_kernel<<<(204864 + 255) / 256, 256, 0, stream>>>(colsum, 204864);
  // 1) row norms (one wave per row, 16384 waves)
  rownorm_kernel<<<2048, 256, 0, stream>>>(X, Y, XX, YY);
  // 2) one-time f32 -> f16 hi/lo splits (and transposed split of X)
  split_xy_kernel<<<8192, 256, 0, stream>>>(X, Y, Xh, Xl, Yh, Yl);
  transpose_split_kernel<<<512, 256, 0, stream>>>(X, XTh, XTl);
  // 3) pd via split-f16 WMMA (16384 wave jobs of 64x64)
  gemm_pd_kernel<<<2048, 256, 0, stream>>>(Xh, Xl, Yh, Yl, XX, YY, PD);
  // 4) median pass 1 (LDS-privatized 64K-bin histogram) + diagonal removal
  hist1_kernel<<<128, 1024, 65536 * sizeof(unsigned), stream>>>(PD, hist1);
  diag1_kernel<<<32, 256, 0, stream>>>(PD, hist1);
  select1_kernel<<<1, 1024, 0, stream>>>(hist1, selu);
  // 5) median pass 2 (low-16 bits inside candidate bins) + diagonal removal
  hist2_kernel<<<2048, 256, 0, stream>>>(PD, selu, hist2);
  diag2_kernel<<<32, 256, 0, stream>>>(PD, selu, hist2);
  select2_kernel<<<1, 1024, 0, stream>>>(hist2, selu, self);
  // 6) K = exp(-gamma*pd) in place, fused column sums
  exp_colsum_kernel<<<128, 256, 0, stream>>>(PD, self, colsum);
  // 7) grad_K = -2*gamma*(K^T X - Y*colsum) via split-f16 WMMA
  grad_gemm_kernel<<<128, 128, 0, stream>>>(PD, XTh, XTl, Y, colsum, self, G);
}
